// cls_model_4690104287378
// MI455X (gfx1250) — matrix operand order repaired and checked by reading, not compiled, not run on hardware
//
#include <hip/hip_runtime.h>
#include <hip/hip_bf16.h>
#include <stdint.h>

// ---------------------------------------------------------------------------
// Reeb-ChebNet forward for MI455X (gfx1250).
// Big GEMMs (Gram 2048x2048x1000, L2@out 2048x1000x2048 x2, xk@W2
// 2048x1000x3000, all x16 batch) run on v_wmma_f32_16x16x32_bf16.
// 128x64 output tile per block, 8 waves, each wave 32x32 (4 WMMA / k-step).
// Main K-loop is branch-free (tail peeled; N-bound handled by clamping,
// harmless since padded C columns are never stored).
// Stage-1 (CIN=6) Laplacian is applied matrix-free. Workspace ~750MB.
// ---------------------------------------------------------------------------

#define BB 16
#define NNP 2048
#define CIN 6
#define RR 20
#define GGS 128
#define F1 1000
#define NCLASS 40

#define LDSP 40  // padded LDS row stride (halfwords): 80B -> aligned, fewer conflicts

typedef __bf16 bf16_t;
typedef __attribute__((ext_vector_type(16))) __bf16 v16bf;
typedef __attribute__((ext_vector_type(8))) float v8f;

union FragU { unsigned int u[8]; v16bf v; };
struct __align__(16) U4 { unsigned int x[4]; };

__device__ inline unsigned short f2bf(float f) {
  unsigned int u = __float_as_uint(f);
  u += 0x7FFFu + ((u >> 16) & 1u);   // round-to-nearest-even
  return (unsigned short)(u >> 16);
}

__device__ inline v8f wmma_bf16(v16bf a, v16bf b, v8f c) {
  return __builtin_amdgcn_wmma_f32_16x16x32_bf16(
      false, a, false, b, (short)0, c, false, false);
}

// A fragment: 16x32 bf16. lane L: m = L&15, half h = L>>4.
// VGPR p(0..3): K pair (2p + 8h); p(4..7): K pair (16 + 2(p-4) + 8h).
__device__ inline v16bf load_frag_a(const unsigned short (*T)[LDSP], int row0, int lane) {
  int m = row0 + (lane & 15);
  int h = (lane >> 4) * 8;
  FragU u;
#pragma unroll
  for (int p = 0; p < 4; ++p) {
    u.u[p]     = *(const unsigned int*)&T[m][2 * p + h];
    u.u[p + 4] = *(const unsigned int*)&T[m][16 + 2 * p + h];
  }
  return u.v;
}

// B fragment: 32x16 bf16 stored in LDS transposed as [n][k].
// lane L: n = L&15, half h = L>>4; VGPR p(0..3): K pair (2p + 8h); p(4..7): K pair (16 + 2(p-4) + 8h).
__device__ inline v16bf load_frag_b(const unsigned short (*T)[LDSP], int col0, int lane) {
  int n = col0 + (lane & 15);
  int h = (lane >> 4) * 8;
  FragU u;
#pragma unroll
  for (int p = 0; p < 4; ++p) {
    u.u[p]     = *(const unsigned int*)&T[n][2 * p + h];
    u.u[p + 4] = *(const unsigned int*)&T[n][16 + 2 * p + h];
  }
  return u.v;
}

// ---------------------------------------------------------------------------
// small kernels
// ---------------------------------------------------------------------------

__global__ void k_init(float* acc) {
  if (threadIdx.x < 8) acc[threadIdx.x] = 0.f;
}

__global__ void k_sqrows6(const float* __restrict__ x, float* __restrict__ sq) {
  int i = blockIdx.x * blockDim.x + threadIdx.x;
  if (i >= BB * NNP) return;
  const float* p = x + (size_t)i * CIN;
  float s = 0.f;
#pragma unroll
  for (int c = 0; c < CIN; ++c) { float v = p[c]; s += v * v; }
  sq[i] = s;
}

__global__ __launch_bounds__(256) void k_rbf_rowsum6(
    const float* __restrict__ x, const float* __restrict__ sq, float* __restrict__ dinv) {
  int row = blockIdx.x, b = row / NNP, i = row % NNP, tid = threadIdx.x;
  const float* xb = x + (size_t)b * NNP * CIN;
  float xi[CIN];
#pragma unroll
  for (int c = 0; c < CIN; ++c) xi[c] = xb[i * CIN + c];
  float sqi = sq[row], s = 0.f;
  for (int j = tid; j < NNP; j += 256) {
    const float* xj = xb + (size_t)j * CIN;
    float d = 0.f;
#pragma unroll
    for (int c = 0; c < CIN; ++c) d += xi[c] * xj[c];
    s += expf(2.f * d - sqi - sq[b * NNP + j]);
  }
  __shared__ float red[256];
  red[tid] = s; __syncthreads();
  for (int st = 128; st > 0; st >>= 1) {
    if (tid < st) red[tid] += red[tid + st];
    __syncthreads();
  }
  if (tid == 0) dinv[row] = rsqrtf(red[0]);
}

// vout = scale*(L vin) - vsub  with L applied matrix-free from x-adjacency
__global__ __launch_bounds__(256) void k_lapply6(
    const float* __restrict__ x, const float* __restrict__ sq, const float* __restrict__ dinv,
    const float* __restrict__ vin, const float* __restrict__ vsub, float scale,
    float* __restrict__ vout) {
  int row = blockIdx.x, b = row / NNP, i = row % NNP, tid = threadIdx.x;
  const float* xb = x + (size_t)b * NNP * CIN;
  const float* vb = vin + (size_t)b * NNP * CIN;
  float xi[CIN];
#pragma unroll
  for (int c = 0; c < CIN; ++c) xi[c] = xb[i * CIN + c];
  float sqi = sq[row], di = dinv[row];
  float acc[CIN] = {0.f, 0.f, 0.f, 0.f, 0.f, 0.f};
  for (int j = tid; j < NNP; j += 256) {
    const float* xj = xb + (size_t)j * CIN;
    float d = 0.f;
#pragma unroll
    for (int c = 0; c < CIN; ++c) d += xi[c] * xj[c];
    float w = expf(2.f * d - sqi - sq[b * NNP + j]) * dinv[b * NNP + j];
#pragma unroll
    for (int c = 0; c < CIN; ++c) acc[c] += w * vb[(size_t)j * CIN + c];
  }
  __shared__ float red[CIN][256];
#pragma unroll
  for (int c = 0; c < CIN; ++c) red[c][tid] = acc[c];
  __syncthreads();
  for (int st = 128; st > 0; st >>= 1) {
    if (tid < st)
#pragma unroll
      for (int c = 0; c < CIN; ++c) red[c][tid] += red[c][tid + st];
    __syncthreads();
  }
  if (tid < CIN) {
    int c = tid;
    float lv = vb[(size_t)i * CIN + c] - di * red[c][0];
    float o = scale * lv - (vsub ? vsub[((size_t)b * NNP + i) * CIN + c] : 0.f);
    vout[((size_t)b * NNP + i) * CIN + c] = o;
  }
}

// out = relu(sum_{c,k} xs_k[.,c] * W1[c*3+k, f] + b1[f]); also bf16 copy
__global__ void k_cheb1(const float* __restrict__ x0, const float* __restrict__ x1,
                        const float* __restrict__ x2, const float* __restrict__ W,
                        const float* __restrict__ bias, float* __restrict__ outF,
                        unsigned short* __restrict__ outBF) {
  long long idx = (long long)blockIdx.x * blockDim.x + threadIdx.x;
  if (idx >= (long long)BB * NNP * F1) return;
  int f = (int)(idx % F1);
  long long bn = idx / F1;
  const float* xs[3] = {x0 + bn * CIN, x1 + bn * CIN, x2 + bn * CIN};
  float s = bias[f];
#pragma unroll
  for (int c = 0; c < CIN; ++c)
#pragma unroll
    for (int k = 0; k < 3; ++k)
      s += xs[k][c] * W[(c * 3 + k) * F1 + f];
  s = fmaxf(s, 0.f);
  outF[idx] = s;
  outBF[idx] = f2bf(s);
}

// accumulate ||out^T y||_F^2 into acc[which]  (y = L@x, shape B,N,6)
__global__ void k_reg(const float* __restrict__ out, const float* __restrict__ y,
                      float* acc, int which) {
  int idx = blockIdx.x * blockDim.x + threadIdx.x;
  if (idx >= BB * F1) return;
  int b = idx / F1, f = idx % F1;
  const float* ob = out + (size_t)b * NNP * F1 + f;
  const float* yb = y + (size_t)b * NNP * CIN;
  float m[CIN] = {0.f, 0.f, 0.f, 0.f, 0.f, 0.f};
  for (int n = 0; n < NNP; ++n) {
    float o = ob[(size_t)n * F1];
#pragma unroll
    for (int c = 0; c < CIN; ++c) m[c] += o * yb[(size_t)n * CIN + c];
  }
  float ss = 0.f;
#pragma unroll
  for (int c = 0; c < CIN; ++c) ss += m[c] * m[c];
  atomicAdd(&acc[which], ss);
}

__global__ void k_gather_max(const int* __restrict__ sccs, const float* __restrict__ out,
                             float* __restrict__ vR) {
  int idx = blockIdx.x * blockDim.x + threadIdx.x;
  if (idx >= BB * RR * F1) return;
  int f = idx % F1, t = idx / F1, r = t % RR, b = t / RR;
  const int* sc = sccs + ((size_t)b * RR + r) * GGS;
  float m = -3.4e38f;
  for (int g = 0; g < GGS; ++g) {
    int n = sc[g];
    m = fmaxf(m, out[((size_t)b * NNP + n) * F1 + f]);
  }
  vR[idx] = m;
}

__global__ __launch_bounds__(256) void k_sqrowsF(const float* __restrict__ out,
                                                 float* __restrict__ sq) {
  int row = blockIdx.x, tid = threadIdx.x;
  const float* p = out + (size_t)row * F1;
  float s = 0.f;
  for (int f = tid; f < F1; f += 256) { float v = p[f]; s += v * v; }
  __shared__ float red[256];
  red[tid] = s; __syncthreads();
  for (int st = 128; st > 0; st >>= 1) {
    if (tid < st) red[tid] += red[tid + st];
    __syncthreads();
  }
  if (tid == 0) sq[row] = red[0];
}

__global__ __launch_bounds__(256) void k_rbf_rowsumW(
    const float* __restrict__ Gm, const float* __restrict__ sq, float* __restrict__ dinv) {
  int row = blockIdx.x, b = row / NNP, tid = threadIdx.x;
  const float* Grow = Gm + (size_t)row * NNP;
  float sqi = sq[row], s = 0.f;
  for (int j = tid; j < NNP; j += 256)
    s += expf(2.f * Grow[j] - sqi - sq[b * NNP + j]);
  __shared__ float red[256];
  red[tid] = s; __syncthreads();
  for (int st = 128; st > 0; st >>= 1) {
    if (tid < st) red[tid] += red[tid + st];
    __syncthreads();
  }
  if (tid == 0) dinv[row] = rsqrtf(red[0]);
}

// write L2 (bf16) row + Lx2 row (= L2 @ x) from Gram row
__global__ __launch_bounds__(256) void k_buildL2(
    const float* __restrict__ Gm, const float* __restrict__ sq, const float* __restrict__ dinv,
    const float* __restrict__ x, unsigned short* __restrict__ L2bf, float* __restrict__ Lx2) {
  int row = blockIdx.x, b = row / NNP, i = row % NNP, tid = threadIdx.x;
  const float* Grow = Gm + (size_t)row * NNP;
  const float* xb = x + (size_t)b * NNP * CIN;
  float sqi = sq[row], di = dinv[row];
  float acc[CIN] = {0.f, 0.f, 0.f, 0.f, 0.f, 0.f};
  for (int j = tid; j < NNP; j += 256) {
    float e = expf(2.f * Grow[j] - sqi - sq[b * NNP + j]);
    float dj = dinv[b * NNP + j];
    float l = ((j == i) ? 1.f : 0.f) - di * e * dj;
    L2bf[(size_t)row * NNP + j] = f2bf(l);
    float w = e * dj;
#pragma unroll
    for (int c = 0; c < CIN; ++c) acc[c] += w * xb[(size_t)j * CIN + c];
  }
  __shared__ float red[CIN][256];
#pragma unroll
  for (int c = 0; c < CIN; ++c) red[c][tid] = acc[c];
  __syncthreads();
  for (int st = 128; st > 0; st >>= 1) {
    if (tid < st)
#pragma unroll
      for (int c = 0; c < CIN; ++c) red[c][tid] += red[c][tid + st];
    __syncthreads();
  }
  if (tid < CIN) {
    int c = tid;
    Lx2[(size_t)row * CIN + c] = xb[(size_t)i * CIN + c] - di * red[c][0];
  }
}

// ---------------------------------------------------------------------------
// WMMA kernels: 128x64 C-tile per 256-thread block, 8 waves x (32x32)
// ---------------------------------------------------------------------------

// Gram: G[b,m,n] = sum_k A[b,m,k]*A[b,n,k]   (A bf16 NNPxF1, G f32 NNPxNNP)
__global__ __launch_bounds__(256) void k_gram_wmma(const unsigned short* __restrict__ Abf,
                                                   float* __restrict__ Gm) {
  __shared__ __align__(16) unsigned short At[128][LDSP];
  __shared__ __align__(16) unsigned short Bt[64][LDSP];
  int b = blockIdx.z;
  int m0 = blockIdx.x * 128, n0 = blockIdx.y * 64;
  const unsigned short* Ab = Abf + (size_t)b * NNP * F1;
  int tid = threadIdx.x, lane = tid & 31, wave = tid >> 5;
  int wm = wave >> 1, wn = wave & 1;
  v8f z = {0.f, 0.f, 0.f, 0.f, 0.f, 0.f, 0.f, 0.f};
  v8f acc00 = z, acc01 = z, acc10 = z, acc11 = z;

  int ar = tid >> 1;             // A row 0..127
  int ac = (tid & 1) * 16;       // A col chunk base (0 or 16)
  int bn = tid >> 2;             // B "n" row 0..63
  int bc = (tid & 3) * 8;        // B col chunk (0,8,16,24)
  const unsigned short* Arow = Ab + (size_t)(m0 + ar) * F1;
  const unsigned short* Brow = Ab + (size_t)(n0 + bn) * F1;

  int k0 = 0;
  // main loop: all chunks fully in-bounds (branch-free)
  for (; k0 < F1 - 31; k0 += 32) {
#pragma unroll
    for (int h = 0; h < 2; ++h)
      *(U4*)&At[ar][ac + h * 8] = *(const U4*)(Arow + k0 + ac + h * 8);
    *(U4*)&Bt[bn][bc] = *(const U4*)(Brow + k0 + bc);
    __syncthreads();
    v16bf a0 = load_frag_a(At, wm * 32, lane);
    v16bf a1 = load_frag_a(At, wm * 32 + 16, lane);
    v16bf b0 = load_frag_b(Bt, wn * 32, lane);
    v16bf b1 = load_frag_b(Bt, wn * 32 + 16, lane);
    acc00 = wmma_bf16(a0, b0, acc00);
    acc01 = wmma_bf16(a0, b1, acc01);
    acc10 = wmma_bf16(a1, b0, acc10);
    acc11 = wmma_bf16(a1, b1, acc11);
    __syncthreads();
  }
  // tail: A zero-filled beyond K; B clamped (A zeros annihilate B garbage)
  if (k0 < F1) {
    const U4 zero4 = {{0u, 0u, 0u, 0u}};
#pragma unroll
    for (int h = 0; h < 2; ++h) {
      int kg = k0 + ac + h * 8;
      U4 v = zero4;
      if (kg < F1) v = *(const U4*)(Arow + kg);
      *(U4*)&At[ar][ac + h * 8] = v;
    }
    {
      int kg = k0 + bc; if (kg > F1 - 8) kg = F1 - 8;
      *(U4*)&Bt[bn][bc] = *(const U4*)(Brow + kg);
    }
    __syncthreads();
    v16bf a0 = load_frag_a(At, wm * 32, lane);
    v16bf a1 = load_frag_a(At, wm * 32 + 16, lane);
    v16bf b0 = load_frag_b(Bt, wn * 32, lane);
    v16bf b1 = load_frag_b(Bt, wn * 32 + 16, lane);
    acc00 = wmma_bf16(a0, b0, acc00);
    acc01 = wmma_bf16(a0, b1, acc01);
    acc10 = wmma_bf16(a1, b0, acc10);
    acc11 = wmma_bf16(a1, b1, acc11);
  }
  int mbase = m0 + wm * 32 + 8 * (lane >> 4);
  int nbase = n0 + wn * 32 + (lane & 15);
  float* Gb = Gm + (size_t)b * NNP * NNP;
  v8f accs[2][2] = {{acc00, acc01}, {acc10, acc11}};
#pragma unroll
  for (int fr = 0; fr < 2; ++fr)
#pragma unroll
    for (int fc = 0; fc < 2; ++fc)
#pragma unroll
      for (int r = 0; r < 8; ++r)
        Gb[(size_t)(mbase + fr * 16 + r) * NNP + (nbase + fc * 16)] = accs[fr][fc][r];
}

// Generic batched NN GEMM, A may be a K-concat of up to 3 pieces (lda==kchunk).
// mode 0: CoutBF = bf16(acc)
// mode 1: CoutBF = bf16(2*acc - aux)          (Chebyshev recurrence fusion)
// mode 2: CoutF  = relu(acc + bias[n])
__global__ __launch_bounds__(256) void k_gemm_bf16(
    const unsigned short* __restrict__ A0, const unsigned short* __restrict__ A1,
    const unsigned short* __restrict__ A2, int kchunk, long long strideA,
    const unsigned short* __restrict__ Bm, int ldb, long long strideB,
    int M, int Ncols, int K, int mode,
    const float* __restrict__ aux, int ldaux, long long strideAux,
    const float* __restrict__ bias,
    unsigned short* __restrict__ CoutBF, float* __restrict__ CoutF,
    int ldc, long long strideC) {
  __shared__ __align__(16) unsigned short At[128][LDSP];
  __shared__ __align__(16) unsigned short Bt[64][LDSP];
  int b = blockIdx.z;
  int m0 = blockIdx.x * 128, n0 = blockIdx.y * 64;
  int tid = threadIdx.x, lane = tid & 31, wave = tid >> 5;
  int wm = wave >> 1, wn = wave & 1;
  v8f z = {0.f, 0.f, 0.f, 0.f, 0.f, 0.f, 0.f, 0.f};
  v8f acc00 = z, acc01 = z, acc10 = z, acc11 = z;

  int ar = tid >> 1;             // A row 0..127
  int ac = (tid & 1) * 16;       // A col chunk base (0 or 16)
  int bkr = tid >> 3;            // B k-row 0..31
  int bnn = (tid & 7) * 8;       // B n chunk (0..56)

  // per-piece row bases for A (chunks never straddle piece seams)
  const unsigned short* Ar0 = A0 + (size_t)b * strideA + (size_t)(m0 + ar) * kchunk;
  const unsigned short* Ar1 = A1 ? A1 + (size_t)b * strideA + (size_t)(m0 + ar) * kchunk : Ar0;
  const unsigned short* Ar2 = A2 ? A2 + (size_t)b * strideA + (size_t)(m0 + ar) * kchunk : Ar0;
  // B column base, clamped: padded C columns are never stored, so garbage OK
  int bn_idx = n0 + bnn; if (bn_idx > Ncols - 8) bn_idx = Ncols - 8;
  const unsigned short* Bp = Bm + (size_t)b * strideB + bn_idx;

  int k0 = 0;
  for (; k0 < K - 31; k0 += 32) {     // branch-free main loop
#pragma unroll
    for (int h = 0; h < 2; ++h) {
      int kg = k0 + ac + h * 8;
      const unsigned short* Ap; int kk;
      if (kg >= 2 * kchunk)      { Ap = Ar2; kk = kg - 2 * kchunk; }
      else if (kg >= kchunk)     { Ap = Ar1; kk = kg - kchunk; }
      else                       { Ap = Ar0; kk = kg; }
      *(U4*)&At[ar][ac + h * 8] = *(const U4*)(Ap + kk);
    }
    {
      U4 v = *(const U4*)(Bp + (size_t)(k0 + bkr) * ldb);
      const unsigned short* vs = (const unsigned short*)&v;
#pragma unroll
      for (int i = 0; i < 8; ++i) Bt[bnn + i][bkr] = vs[i];
    }
    __syncthreads();
    v16bf a0 = load_frag_a(At, wm * 32, lane);
    v16bf a1 = load_frag_a(At, wm * 32 + 16, lane);
    v16bf b0 = load_frag_b(Bt, wn * 32, lane);
    v16bf b1 = load_frag_b(Bt, wn * 32 + 16, lane);
    acc00 = wmma_bf16(a0, b0, acc00);
    acc01 = wmma_bf16(a0, b1, acc01);
    acc10 = wmma_bf16(a1, b0, acc10);
    acc11 = wmma_bf16(a1, b1, acc11);
    __syncthreads();
  }
  if (k0 < K) {                       // tail: A zero-guarded, B k clamped
    const U4 zero4 = {{0u, 0u, 0u, 0u}};
#pragma unroll
    for (int h = 0; h < 2; ++h) {
      int kg = k0 + ac + h * 8;
      U4 v = zero4;
      if (kg < K) {
        const unsigned short* Ap; int kk;
        if (kg >= 2 * kchunk)      { Ap = Ar2; kk = kg - 2 * kchunk; }
        else if (kg >= kchunk)     { Ap = Ar1; kk = kg - kchunk; }
        else                       { Ap = Ar0; kk = kg; }
        v = *(const U4*)(Ap + kk);
      }
      *(U4*)&At[ar][ac + h * 8] = v;
    }
    {
      int kg = k0 + bkr; if (kg > K - 1) kg = K - 1;
      U4 v = *(const U4*)(Bp + (size_t)kg * ldb);
      const unsigned short* vs = (const unsigned short*)&v;
#pragma unroll
      for (int i = 0; i < 8; ++i) Bt[bnn + i][bkr] = vs[i];
    }
    __syncthreads();
    v16bf a0 = load_frag_a(At, wm * 32, lane);
    v16bf a1 = load_frag_a(At, wm * 32 + 16, lane);
    v16bf b0 = load_frag_b(Bt, wn * 32, lane);
    v16bf b1 = load_frag_b(Bt, wn * 32 + 16, lane);
    acc00 = wmma_bf16(a0, b0, acc00);
    acc01 = wmma_bf16(a0, b1, acc01);
    acc10 = wmma_bf16(a1, b0, acc10);
    acc11 = wmma_bf16(a1, b1, acc11);
  }
  int mbase = m0 + wm * 32 + 8 * (lane >> 4);
  int nbase = n0 + wn * 32 + (lane & 15);
  v8f accs[2][2] = {{acc00, acc01}, {acc10, acc11}};
#pragma unroll
  for (int fr = 0; fr < 2; ++fr)
#pragma unroll
    for (int fc = 0; fc < 2; ++fc)
#pragma unroll
      for (int r = 0; r < 8; ++r) {
        int m = mbase + fr * 16 + r;
        int n = nbase + fc * 16;
        if (m >= M || n >= Ncols) continue;
        float v = accs[fr][fc][r];
        size_t off = (size_t)b * strideC + (size_t)m * ldc + n;
        if (mode == 0) {
          CoutBF[off] = f2bf(v);
        } else if (mode == 1) {
          float a = aux[(size_t)b * strideAux + (size_t)m * ldaux + n];
          CoutBF[off] = f2bf(2.f * v - a);
        } else {
          CoutF[off] = fmaxf(v + bias[n], 0.f);
        }
      }
}

// ---------------------------------------------------------------------------
// Reeb path + FC head + scalars
// ---------------------------------------------------------------------------

__global__ void k_w2bf(const float* __restrict__ W2, unsigned short* __restrict__ w2bf) {
  long long idx = (long long)blockIdx.x * blockDim.x + threadIdx.x;
  if (idx >= (long long)3 * F1 * F1) return;
  int f = (int)(idx % F1);
  int row = (int)(idx / F1);        // row = c*3 + k
  int c = row / 3, k = row % 3;
  // reorder rows to piece-major [k*F1 + c] to match K-concat [out|t1|t2]
  w2bf[((size_t)(k * F1 + c)) * F1 + f] = f2bf(W2[idx]);
}

__global__ void k_reeb_step(const float* __restrict__ LR, const float* __restrict__ vin,
                            const float* __restrict__ vsub, float scale,
                            float* __restrict__ vout) {
  int idx = blockIdx.x * blockDim.x + threadIdx.x;
  if (idx >= BB * RR * F1) return;
  int f = idx % F1, t = idx / F1, r = t % RR, b = t / RR;
  const float* Lrow = LR + ((size_t)b * RR + r) * RR;
  const float* vb = vin + (size_t)b * RR * F1;
  float s = 0.f;
#pragma unroll
  for (int q = 0; q < RR; ++q) s += Lrow[q] * vb[(size_t)q * F1 + f];
  vout[idx] = scale * s - (vsub ? vsub[idx] : 0.f);
}

__global__ void k_reeb_gemm(const float* __restrict__ x0, const float* __restrict__ x1,
                            const float* __restrict__ x2, const float* __restrict__ x3,
                            const float* __restrict__ x4, const float* __restrict__ x5,
                            const float* __restrict__ Wr, const float* __restrict__ br,
                            float* __restrict__ outR) {
  int idx = blockIdx.x * blockDim.x + threadIdx.x;
  if (idx >= BB * RR * F1) return;
  int f = idx % F1, t = idx / F1, r = t % RR, b = t / RR;
  size_t base = ((size_t)b * RR + r) * F1;
  const float* xs[6] = {x0 + base, x1 + base, x2 + base, x3 + base, x4 + base, x5 + base};
  float s = br[f];
  for (int c = 0; c < F1; ++c) {
#pragma unroll
    for (int k = 0; k < 6; ++k) s += xs[k][c] * Wr[(size_t)(c * 6 + k) * F1 + f];
  }
  outR[idx] = fmaxf(s, 0.f);
}

__global__ void k_colmax(const float* __restrict__ src, int rows, float* __restrict__ dst) {
  int idx = blockIdx.x * blockDim.x + threadIdx.x;
  if (idx >= BB * F1) return;
  int b = idx / F1, f = idx % F1;
  const float* p = src + (size_t)b * rows * F1 + f;
  float m = p[0];
  for (int n = 1; n < rows; ++n) m = fmaxf(m, p[(size_t)n * F1]);
  dst[idx] = m;
}

__global__ void k_fc(const float* __restrict__ in1, int n1, const float* __restrict__ in2,
                     int n2, const float* __restrict__ W, const float* __restrict__ bias,
                     int nout, int relu, float* __restrict__ out) {
  int idx = blockIdx.x * blockDim.x + threadIdx.x;
  if (idx >= BB * nout) return;
  int b = idx / nout, o = idx % nout;
  float s = bias[o];
  for (int i = 0; i < n1; ++i) s += in1[(size_t)b * n1 + i] * W[(size_t)i * nout + o];
  for (int i = 0; i < n2; ++i) s += in2[(size_t)b * n2 + i] * W[(size_t)(n1 + i) * nout + o];
  if (relu) s = fmaxf(s, 0.f);
  out[idx] = s;
}

__global__ __launch_bounds__(256) void k_scalars(
    const float* __restrict__ acc, const float* __restrict__ Wfc1, const float* __restrict__ bfc1,
    const float* __restrict__ Wfc2, const float* __restrict__ bfc2,
    const float* __restrict__ Wfc3, const float* __restrict__ bfc3, float* __restrict__ outreg) {
  __shared__ float red[256];
  int tid = threadIdx.x;
  float s = 0.f;
  for (int i = tid; i < 2 * F1; i += 256) { float w = Wfc1[(size_t)i * 512]; s += w * w; }
  red[tid] = s; __syncthreads();
  for (int st = 128; st > 0; st >>= 1) { if (tid < st) red[tid] += red[tid + st]; __syncthreads(); }
  if (tid == 0) outreg[2] = red[0];
  __syncthreads();
  s = 0.f;
  for (int i = tid; i < 512; i += 256) { float w = Wfc2[(size_t)i * 128]; s += w * w; }
  red[tid] = s; __syncthreads();
  for (int st = 128; st > 0; st >>= 1) { if (tid < st) red[tid] += red[tid + st]; __syncthreads(); }
  if (tid == 0) outreg[4] = red[0];
  __syncthreads();
  s = 0.f;
  for (int i = tid; i < 128; i += 256) { float w = Wfc3[(size_t)i * NCLASS]; s += w * w; }
  red[tid] = s; __syncthreads();
  for (int st = 128; st > 0; st >>= 1) { if (tid < st) red[tid] += red[tid + st]; __syncthreads(); }
  if (tid == 0) {
    outreg[6] = red[0];
    outreg[0] = sqrtf(acc[0]);
    outreg[1] = sqrtf(acc[1]);
    outreg[3] = bfc1[0] * bfc1[0];
    outreg[5] = bfc2[0] * bfc2[0];
    outreg[7] = bfc3[0] * bfc3[0];
  }
}

// ---------------------------------------------------------------------------
// launcher
// ---------------------------------------------------------------------------

extern "C" void kernel_launch(void* const* d_in, const int* in_sizes, int n_in,
                              void* d_out, int out_size, void* d_ws, size_t ws_size,
                              hipStream_t stream) {
  const float* x   = (const float*)d_in[0];
  const float* LR  = (const float*)d_in[1];
  const int*   scc = (const int*)d_in[2];
  const float* W1  = (const float*)d_in[3];
  const float* b1  = (const float*)d_in[4];
  const float* W2  = (const float*)d_in[5];
  const float* b2  = (const float*)d_in[6];
  const float* Wr  = (const float*)d_in[7];
  const float* br  = (const float*)d_in[8];
  const float* Wfc1 = (const float*)d_in[9];
  const float* bfc1 = (const float*)d_in[10];
  const float* Wfc2 = (const float*)d_in[11];
  const float* bfc2 = (const float*)d_in[12];
  const float* Wfc3 = (const float*)d_in[13];
  const float* bfc3 = (const float*)d_in[14];
  float* outp = (float*)d_out;

  char* W = (char*)d_ws;
  size_t off = 0;
  auto alloc = [&](size_t bytes) { size_t o = off; off += (bytes + 255) & ~(size_t)255; return o; };

  float*          ACC   = (float*)(W + alloc(64));
  float*          SQX   = (float*)(W + alloc((size_t)BB * NNP * 4));
  float*          DINV1 = (float*)(W + alloc((size_t)BB * NNP * 4));
  float*          X1    = (float*)(W + alloc((size_t)BB * NNP * CIN * 4));
  float*          X2    = (float*)(W + alloc((size_t)BB * NNP * CIN * 4));
  float*          OUTF  = (float*)(W + alloc((size_t)BB * NNP * F1 * 4));
  unsigned short* OUTBF = (unsigned short*)(W + alloc((size_t)BB * NNP * F1 * 2));
  float*          SQO   = (float*)(W + alloc((size_t)BB * NNP * 4));
  float*          DINV2 = (float*)(W + alloc((size_t)BB * NNP * 4));
  float*          GM    = (float*)(W + alloc((size_t)BB * NNP * NNP * 4)); // reused as out2
  unsigned short* L2BF  = (unsigned short*)(W + alloc((size_t)BB * NNP * NNP * 2));
  float*          LX2   = (float*)(W + alloc((size_t)BB * NNP * CIN * 4));
  unsigned short* T1BF  = (unsigned short*)(W + alloc((size_t)BB * NNP * F1 * 2));
  unsigned short* T2BF  = (unsigned short*)(W + alloc((size_t)BB * NNP * F1 * 2));
  unsigned short* W2BF  = (unsigned short*)(W + alloc((size_t)3 * F1 * F1 * 2));
  float*          VR    = (float*)(W + alloc((size_t)BB * RR * F1 * 4));
  float*          XR1   = (float*)(W + alloc((size_t)BB * RR * F1 * 4));
  float*          XR2   = (float*)(W + alloc((size_t)BB * RR * F1 * 4));
  float*          XR3   = (float*)(W + alloc((size_t)BB * RR * F1 * 4));
  float*          XR4   = (float*)(W + alloc((size_t)BB * RR * F1 * 4));
  float*          XR5   = (float*)(W + alloc((size_t)BB * RR * F1 * 4));
  float*          OREEB = (float*)(W + alloc((size_t)BB * RR * F1 * 4));
  float*          GMAX  = (float*)(W + alloc((size_t)BB * F1 * 4));
  float*          GRMAX = (float*)(W + alloc((size_t)BB * F1 * 4));
  float*          H1    = (float*)(W + alloc((size_t)BB * 512 * 4));
  float*          H2    = (float*)(W + alloc((size_t)BB * 128 * 4));
  float*          OUT2F = GM;  // Gram buffer dead by then

  const int TPB = 256;
  long long nBNF = (long long)BB * NNP * F1;
  int gBNF = (int)((nBNF + TPB - 1) / TPB);
  int gRowsBNF1 = (BB * F1 + TPB - 1) / TPB;
  int gReeb = (BB * RR * F1 + TPB - 1) / TPB;

  k_init<<<1, 32, 0, stream>>>(ACC);

  // ---- stage 1: matrix-free Laplacian + Cheb(K=3) ----
  k_sqrows6<<<(BB * NNP + TPB - 1) / TPB, TPB, 0, stream>>>(x, SQX);
  k_rbf_rowsum6<<<BB * NNP, TPB, 0, stream>>>(x, SQX, DINV1);
  k_lapply6<<<BB * NNP, TPB, 0, stream>>>(x, SQX, DINV1, x,  nullptr, 1.f, X1);
  k_lapply6<<<BB * NNP, TPB, 0, stream>>>(x, SQX, DINV1, X1, x,       2.f, X2);
  k_cheb1<<<gBNF, TPB, 0, stream>>>(x, X1, X2, W1, b1, OUTF, OUTBF);
  k_reg<<<gRowsBNF1, TPB, 0, stream>>>(OUTF, X1, ACC, 0);
  k_gather_max<<<gReeb, TPB, 0, stream>>>(scc, OUTF, VR);

  // ---- stage 2: WMMA Gram -> L2 -> Cheb(K=3) GEMMs ----
  k_sqrowsF<<<BB * NNP, TPB, 0, stream>>>(OUTF, SQO);
  k_gram_wmma<<<dim3(NNP / 128, NNP / 64, BB), TPB, 0, stream>>>(OUTBF, GM);
  k_rbf_rowsumW<<<BB * NNP, TPB, 0, stream>>>(GM, SQO, DINV2);
  k_buildL2<<<BB * NNP, TPB, 0, stream>>>(GM, SQO, DINV2, x, L2BF, LX2);

  int nTiles = (F1 + 63) / 64;
  // t1 = L2 @ out                       -> bf16
  k_gemm_bf16<<<dim3(NNP / 128, nTiles, BB), TPB, 0, stream>>>(
      L2BF, nullptr, nullptr, NNP, (long long)NNP * NNP,
      OUTBF, F1, (long long)NNP * F1,
      NNP, F1, NNP, 0, nullptr, 0, 0, nullptr,
      T1BF, nullptr, F1, (long long)NNP * F1);
  // t2 = 2*(L2 @ t1) - out              -> bf16 (fused Chebyshev recurrence)
  k_gemm_bf16<<<dim3(NNP / 128, nTiles, BB), TPB, 0, stream>>>(
      L2BF, nullptr, nullptr, NNP, (long long)NNP * NNP,
      T1BF, F1, (long long)NNP * F1,
      NNP, F1, NNP, 1, OUTF, F1, (long long)NNP * F1, nullptr,
      nullptr, nullptr, F1, (long long)NNP * F1);
  // out2 = relu([out|t1|t2] @ W2 + b2)  -> f32 (K = 3000 concat)
  k_w2bf<<<(int)(((long long)3 * F1 * F1 + TPB - 1) / TPB), TPB, 0, stream>>>(W2, W2BF);
  k_gemm_bf16<<<dim3(NNP / 128, nTiles, BB), TPB, 0, stream>>>(
      OUTBF, T1BF, T2BF, F1, (long long)NNP * F1,
      W2BF, F1, 0LL,
      NNP, F1, 3 * F1, 2, nullptr, 0, 0, b2,
      nullptr, OUT2F, F1, (long long)NNP * F1);

  k_reg<<<gRowsBNF1, TPB, 0, stream>>>(OUT2F, LX2, ACC, 1);
  k_colmax<<<gRowsBNF1, TPB, 0, stream>>>(OUT2F, NNP, GMAX);

  // ---- Reeb path: Cheb(K=6) on 20-node graph ----
  k_reeb_step<<<gReeb, TPB, 0, stream>>>(LR, VR,  nullptr, 1.f, XR1);
  k_reeb_step<<<gReeb, TPB, 0, stream>>>(LR, XR1, VR,      2.f, XR2);
  k_reeb_step<<<gReeb, TPB, 0, stream>>>(LR, XR2, XR1,     2.f, XR3);
  k_reeb_step<<<gReeb, TPB, 0, stream>>>(LR, XR3, XR2,     2.f, XR4);
  k_reeb_step<<<gReeb, TPB, 0, stream>>>(LR, XR4, XR3,     2.f, XR5);
  k_reeb_gemm<<<gReeb, TPB, 0, stream>>>(VR, XR1, XR2, XR3, XR4, XR5, Wr, br, OREEB);
  k_colmax<<<gRowsBNF1, TPB, 0, stream>>>(OREEB, RR, GRMAX);

  // ---- FC head ----
  k_fc<<<(BB * 512 + TPB - 1) / TPB, TPB, 0, stream>>>(GRMAX, F1, GMAX, F1, Wfc1, bfc1, 512, 1, H1);
  k_fc<<<(BB * 128 + TPB - 1) / TPB, TPB, 0, stream>>>(H1, 512, nullptr, 0, Wfc2, bfc2, 128, 1, H2);
  k_fc<<<(BB * NCLASS + TPB - 1) / TPB, TPB, 0, stream>>>(H2, 128, nullptr, 0, Wfc3, bfc3, NCLASS, 0, outp);

  // ---- regularizers ----
  k_scalars<<<1, TPB, 0, stream>>>(ACC, Wfc1, bfc1, Wfc2, bfc2, Wfc3, bfc3, outp + BB * NCLASS);

  (void)in_sizes; (void)n_in; (void)out_size; (void)ws_size;
}
